// HeteroGNNEncoder_82033875353615
// MI455X (gfx1250) — matrix operand order repaired and checked by reading, not compiled, not run on hardware
//
#include <hip/hip_runtime.h>
#include <stdint.h>
#include <stddef.h>

// ---------------------------------------------------------------------------
// Types for CDNA5 WMMA
// ---------------------------------------------------------------------------
typedef __bf16          v16bf __attribute__((ext_vector_type(16)));
typedef unsigned short  v16us __attribute__((ext_vector_type(16)));
typedef float           v8f   __attribute__((ext_vector_type(8)));

// ---------------------------------------------------------------------------
// Small device helpers
// ---------------------------------------------------------------------------
__device__ __forceinline__ unsigned short f2bf(float f) {
    unsigned u = __float_as_uint(f);
    unsigned r = u + 0x7fffu + ((u >> 16) & 1u);   // round-to-nearest-even
    return (unsigned short)(r >> 16);
}
__device__ __forceinline__ float bf2f(unsigned short s) {
    return __uint_as_float(((unsigned)s) << 16);
}
// order-preserving float -> uint mapping (for atomicMax on floats)
__device__ __forceinline__ unsigned fmapf(float f) {
    unsigned u = __float_as_uint(f);
    return (u & 0x80000000u) ? ~u : (u | 0x80000000u);
}
__device__ __forceinline__ float funmapf(unsigned u) {
    return __uint_as_float((u & 0x80000000u) ? (u & 0x7fffffffu) : ~u);
}
__device__ __forceinline__ float lrelu(float v) { return v > 0.f ? v : 0.2f * v; }

// ---------------------------------------------------------------------------
// Utility kernels
// ---------------------------------------------------------------------------
__global__ void fill_f32_k(float* p, float v, size_t n) {
    size_t i = (size_t)blockIdx.x * blockDim.x + threadIdx.x;
    if (i < n) p[i] = v;
}
__global__ void fill_u32_k(unsigned* p, unsigned v, size_t n) {
    size_t i = (size_t)blockIdx.x * blockDim.x + threadIdx.x;
    if (i < n) p[i] = v;
}
// x [N,K] f32  ->  y [N,KP] bf16, zero-padded K..KP-1
__global__ void to_bf16_pad_k(const float* __restrict__ x, unsigned short* __restrict__ y,
                              int K, int KP, size_t total) {
    size_t i = (size_t)blockIdx.x * blockDim.x + threadIdx.x;
    if (i >= total) return;
    size_t n = i / KP;
    int    k = (int)(i - n * KP);
    y[i] = (k < K) ? f2bf(x[n * K + k]) : (unsigned short)0;
}
// W [K,N] f32 row-major  ->  Wt [N,KP] bf16 (B operand, K-contiguous, zero-padded)
__global__ void transpose_pad_bf16_k(const float* __restrict__ W, unsigned short* __restrict__ Wt,
                                     int K, int N, int KP) {
    int t = blockIdx.x * blockDim.x + threadIdx.x;
    if (t >= N * KP) return;
    int n = t / KP, k = t - n * KP;
    Wt[t] = (k < K) ? f2bf(W[(size_t)k * N + n]) : (unsigned short)0;
}

// ---------------------------------------------------------------------------
// bf16 WMMA GEMM:  C[M,N] (+)= A[M,KP]bf16 @ B[KP,N]  (B passed transposed [N,KP])
// KP is a compile-time multiple of 32 (staging buffers zero-padded), so all
// fragment loads are unconditional b128s and the K loop fully unrolls.
// One wave per 16x16 tile. M,N multiples of 16.
// ---------------------------------------------------------------------------
union FragU { uint4 q[2]; v16us v; };

template <int KP>
__global__ __launch_bounds__(256)
void gemm_bf16_wmma(const unsigned short* __restrict__ A,
                    const unsigned short* __restrict__ Bt,
                    const float* __restrict__ bias,
                    float* __restrict__ Cf,
                    unsigned short* __restrict__ Cbf,
                    int M, int N, int accumulate) {
    const int wave = threadIdx.x >> 5;
    const int lane = threadIdx.x & 31;
    const int mt   = blockIdx.x * (blockDim.x >> 5) + wave;
    const int m0   = mt << 4;
    if (m0 >= M) return;                       // uniform per wave
    const int n0   = (int)blockIdx.y << 4;
    const int half = lane >> 4;                // 0 or 1
    const int l15  = lane & 15;

    // ISA 16-bit fragment layouts:
    //  A 16x32: lanes 0-15 -> K 0..7 & 16..23 ; lanes 16-31 -> K 8..15 & 24..31
    //  B 32x16: lanes 0-15 -> K 0..7 & 16..23 ; lanes 16-31 -> K 8..15 & 24..31
    const unsigned short* ap = A  + (size_t)(m0 + l15) * KP + half * 8;
    const unsigned short* bp = Bt + (size_t)(n0 + l15) * KP + half * 8;
    const int ccol = n0 + l15;

    v8f c = {};
    if (accumulate) {
#pragma unroll
        for (int r = 0; r < 8; ++r)
            c[r] = Cf[(size_t)(m0 + r + half * 8) * N + ccol];
    }

#pragma unroll
    for (int kc = 0; kc < KP; kc += 32) {
        FragU fa, fb;
        fa.q[0] = *(const uint4*)(ap + kc);
        fa.q[1] = *(const uint4*)(ap + kc + 16);
        fb.q[0] = *(const uint4*)(bp + kc);
        fb.q[1] = *(const uint4*)(bp + kc + 16);
        v16bf a = __builtin_bit_cast(v16bf, fa.v);
        v16bf b = __builtin_bit_cast(v16bf, fb.v);
        c = __builtin_amdgcn_wmma_f32_16x16x32_bf16(false, a, false, b,
                                                    (short)0, c, false, false);
    }

    if (bias) {
        float bv = bias[ccol];
#pragma unroll
        for (int r = 0; r < 8; ++r) c[r] += bv;
    }
    if (Cbf) {
#pragma unroll
        for (int r = 0; r < 8; ++r)
            Cbf[(size_t)(m0 + r + half * 8) * N + ccol] = f2bf(c[r]);
    } else {
#pragma unroll
        for (int r = 0; r < 8; ++r)
            Cf[(size_t)(m0 + r + half * 8) * N + ccol] = c[r];
    }
}

// ---------------------------------------------------------------------------
// Segment mean (SAGE aggregation) — one thread per (edge, feature)
// ---------------------------------------------------------------------------
__global__ void seg_sum_k(const float* __restrict__ xs, const int* __restrict__ src,
                          const int* __restrict__ dst, float* __restrict__ agg,
                          float* __restrict__ cnt, int E, int K) {
    size_t t = (size_t)blockIdx.x * blockDim.x + threadIdx.x;
    if (t >= (size_t)E * K) return;
    int e = (int)(t / K), j = (int)(t - (size_t)e * K);
    int s = src[e], d = dst[e];
    atomicAdd(&agg[(size_t)d * K + j], xs[(size_t)s * K + j]);
    if (j == 0) atomicAdd(&cnt[d], 1.f);
}
__global__ void seg_div_k(float* __restrict__ agg, const float* __restrict__ cnt,
                          int Ndst, int K) {
    size_t i = (size_t)blockIdx.x * blockDim.x + threadIdx.x;
    if (i >= (size_t)Ndst * K) return;
    agg[i] /= fmaxf(cnt[i / K], 1.f);
}

// ---------------------------------------------------------------------------
// GAT: fold att vectors into [din,4] effective matrices; per-node scores;
// 3-pass edge softmax (max / sum / apply) with self loops.
// ---------------------------------------------------------------------------
__global__ void att_eff_k(const float* __restrict__ W, const float* __restrict__ att,
                          float* __restrict__ eff, int din) {
    int t = blockIdx.x * blockDim.x + threadIdx.x;
    if (t >= din * 4) return;
    int j = t >> 2, h = t & 3;
    float s = 0.f;
    for (int k = 0; k < 64; ++k) s += W[(size_t)j * 256 + h * 64 + k] * att[h * 64 + k];
    eff[j * 4 + h] = s;
}
__global__ void gat_scores_k(const float* __restrict__ x, const float* __restrict__ we,
                             const float* __restrict__ wd, float* __restrict__ a_s,
                             float* __restrict__ a_d, int N, int din) {
    int i = blockIdx.x * blockDim.x + threadIdx.x;
    if (i >= N) return;
    const float* xr = x + (size_t)i * din;
    for (int h = 0; h < 4; ++h) {
        float s = 0.f, d = 0.f;
        for (int j = 0; j < din; ++j) { float v = xr[j]; s += v * we[j * 4 + h]; d += v * wd[j * 4 + h]; }
        a_s[i * 4 + h] = s;
        a_d[i * 4 + h] = d;
    }
}
__global__ void gat_edge_max_k(const float* __restrict__ a_s, const float* __restrict__ a_d,
                               const int* __restrict__ src, const int* __restrict__ dst,
                               unsigned* __restrict__ mmax, int E) {
    int e = blockIdx.x * blockDim.x + threadIdx.x;
    if (e >= E) return;
    int s = src[e], d = dst[e];
    for (int h = 0; h < 4; ++h) {
        float v = lrelu(a_s[s * 4 + h] + a_d[d * 4 + h]);
        atomicMax(&mmax[d * 4 + h], fmapf(v));
    }
}
__global__ void gat_self_max_k(const float* __restrict__ a_s, const float* __restrict__ a_d,
                               unsigned* __restrict__ mmax, int N) {
    int i = blockIdx.x * blockDim.x + threadIdx.x;
    if (i >= N) return;
    for (int h = 0; h < 4; ++h) {
        unsigned m = fmapf(lrelu(a_s[i * 4 + h] + a_d[i * 4 + h]));
        if (m > mmax[i * 4 + h]) mmax[i * 4 + h] = m;
    }
}
__global__ void gat_edge_sum_k(const float* __restrict__ a_s, const float* __restrict__ a_d,
                               const int* __restrict__ src, const int* __restrict__ dst,
                               const unsigned* __restrict__ mmax, float* __restrict__ denom, int E) {
    int e = blockIdx.x * blockDim.x + threadIdx.x;
    if (e >= E) return;
    int s = src[e], d = dst[e];
    for (int h = 0; h < 4; ++h) {
        float v = lrelu(a_s[s * 4 + h] + a_d[d * 4 + h]);
        atomicAdd(&denom[d * 4 + h], expf(v - funmapf(mmax[d * 4 + h])));
    }
}
__global__ void gat_self_sum_k(const float* __restrict__ a_s, const float* __restrict__ a_d,
                               const unsigned* __restrict__ mmax, float* __restrict__ denom, int N) {
    int i = blockIdx.x * blockDim.x + threadIdx.x;
    if (i >= N) return;
    for (int h = 0; h < 4; ++h) {
        float v = lrelu(a_s[i * 4 + h] + a_d[i * 4 + h]);
        denom[i * 4 + h] += expf(v - funmapf(mmax[i * 4 + h]));
    }
}
// precompute per-edge softmax weights (already scaled by 1/HEADS)
__global__ void gat_edge_w_k(const float* __restrict__ a_s, const float* __restrict__ a_d,
                             const int* __restrict__ src, const int* __restrict__ dst,
                             const unsigned* __restrict__ mmax, const float* __restrict__ denom,
                             float* __restrict__ walpha, int E) {
    int e = blockIdx.x * blockDim.x + threadIdx.x;
    if (e >= E) return;
    int s = src[e], d = dst[e];
    for (int h = 0; h < 4; ++h) {
        float v = lrelu(a_s[s * 4 + h] + a_d[d * 4 + h]);
        walpha[(size_t)e * 4 + h] = 0.25f * expf(v - funmapf(mmax[d * 4 + h])) / denom[d * 4 + h];
    }
}
// one thread per (edge, feature)
__global__ void gat_edge_acc_k(const float* __restrict__ walpha,
                               const int* __restrict__ src, const int* __restrict__ dst,
                               const unsigned short* __restrict__ hs, float* __restrict__ acc, int E) {
    size_t t = (size_t)blockIdx.x * blockDim.x + threadIdx.x;
    if (t >= (size_t)E * 64) return;
    int e = (int)(t >> 6), j = (int)(t & 63);
    int s = src[e], d = dst[e];
    const float* w = walpha + (size_t)e * 4;
    const unsigned short* hr = hs + (size_t)s * 256;
    float val = w[0] * bf2f(hr[j]) + w[1] * bf2f(hr[64 + j]) +
                w[2] * bf2f(hr[128 + j]) + w[3] * bf2f(hr[192 + j]);
    atomicAdd(&acc[(size_t)d * 64 + j], val);
}
__global__ void gat_self_acc_k(const float* __restrict__ a_s, const float* __restrict__ a_d,
                               const unsigned* __restrict__ mmax, const float* __restrict__ denom,
                               const unsigned short* __restrict__ hs, const float* __restrict__ bias,
                               float* __restrict__ acc, int N) {
    int i = blockIdx.x * blockDim.x + threadIdx.x;
    if (i >= N) return;
    float w[4];
    for (int h = 0; h < 4; ++h) {
        float v = lrelu(a_s[i * 4 + h] + a_d[i * 4 + h]);
        w[h] = 0.25f * expf(v - funmapf(mmax[i * 4 + h])) / denom[i * 4 + h];
    }
    const unsigned short* hr = hs + (size_t)i * 256;
    float* ar = acc + (size_t)i * 64;
    for (int j = 0; j < 64; ++j) {
        float val = w[0] * bf2f(hr[j]) + w[1] * bf2f(hr[64 + j]) +
                    w[2] * bf2f(hr[128 + j]) + w[3] * bf2f(hr[192 + j]);
        ar[j] += val + bias[j];
    }
}

// ---------------------------------------------------------------------------
// GCN: deg includes self loop (deg prefilled with 1); norm = rsqrt(deg_s*deg_d)
// ---------------------------------------------------------------------------
__global__ void deg_count_k(const int* __restrict__ dst, float* __restrict__ deg, int E) {
    int e = blockIdx.x * blockDim.x + threadIdx.x;
    if (e >= E) return;
    atomicAdd(&deg[dst[e]], 1.f);
}
// one thread per (edge, feature)
__global__ void gcn_edge_k(const float* __restrict__ h, const float* __restrict__ deg,
                           const int* __restrict__ src, const int* __restrict__ dst,
                           float* __restrict__ acc, int E) {
    size_t t = (size_t)blockIdx.x * blockDim.x + threadIdx.x;
    if (t >= (size_t)E * 64) return;
    int e = (int)(t >> 6), j = (int)(t & 63);
    int s = src[e], d = dst[e];
    float nrm = rsqrtf(deg[s]) * rsqrtf(deg[d]);
    atomicAdd(&acc[(size_t)d * 64 + j], nrm * h[(size_t)s * 64 + j]);
}
__global__ void gcn_self_k(const float* __restrict__ h, const float* __restrict__ deg,
                           const float* __restrict__ bias, float* __restrict__ acc, int N) {
    int i = blockIdx.x * blockDim.x + threadIdx.x;
    if (i >= N) return;
    float inv = 1.f / deg[i];                  // dinv[i]^2
    const float* hr = h + (size_t)i * 64;
    float* ar = acc + (size_t)i * 64;
    for (int j = 0; j < 64; ++j) ar[j] += inv * hr[j] + bias[j];
}

// ---------------------------------------------------------------------------
// BatchNorm (batch stats, gamma=1, beta=0) + ReLU over [N,64]
// ---------------------------------------------------------------------------
__global__ void bn_reduce_k(const float* __restrict__ x, float* __restrict__ stats, int N) {
    __shared__ float ssum[256], ssq[256];
    int col = threadIdx.x & 63, q = threadIdx.x >> 6;
    float s = 0.f, s2 = 0.f;
    for (int r = blockIdx.x * 4 + q; r < N; r += gridDim.x * 4) {
        float v = x[(size_t)r * 64 + col];
        s += v; s2 += v * v;
    }
    ssum[threadIdx.x] = s; ssq[threadIdx.x] = s2;
    __syncthreads();
    if (q == 0) {
        for (int k = 1; k < 4; ++k) { s += ssum[col + k * 64]; s2 += ssq[col + k * 64]; }
        atomicAdd(&stats[col], s);
        atomicAdd(&stats[64 + col], s2);
    }
}
__global__ void bn_apply_k(const float* __restrict__ x, const float* __restrict__ stats,
                           float* __restrict__ out, int N) {
    size_t i = (size_t)blockIdx.x * blockDim.x + threadIdx.x;
    if (i >= (size_t)N * 64) return;
    int col = (int)(i & 63);
    float invn = 1.f / (float)N;
    float mu = stats[col] * invn;
    float var = stats[64 + col] * invn - mu * mu;
    float y = (x[i] - mu) * rsqrtf(var + 1e-5f);
    out[i] = fmaxf(y, 0.f);
}

// ---------------------------------------------------------------------------
// Host orchestration
// ---------------------------------------------------------------------------
static inline unsigned gdiv(size_t n, int b) { return (unsigned)((n + b - 1) / b); }
static inline int kpad(int k) { return (k <= 32) ? 32 : 64; }

struct LayerPtr {
    const float *mu_Wl, *mu_bl, *mu_Wr, *du_Wl, *du_bl, *du_Wr;
    const float *um_Wl, *um_bl, *um_Wr, *ud_Wl, *ud_bl, *ud_Wr;
    const float *gat_Wsrc, *gat_Wdst, *gat_as, *gat_ad, *gat_b;
    const float *gcn_W, *gcn_b;
};

extern "C" void kernel_launch(void* const* d_in, const int* in_sizes, int n_in,
                              void* d_out, int out_size, void* d_ws, size_t ws_size,
                              hipStream_t stream) {
    (void)n_in; (void)out_size; (void)ws_size;
    const int NU = 200000, NM = 50000, ND = 100000, E = 1000000;

    const float* x_user     = (const float*)d_in[0];
    const float* x_merchant = (const float*)d_in[1];
    const float* x_device   = (const float*)d_in[2];

    // Param leaf order: jax pytree sorts dict keys; detect insertion-order
    // flattening via in_sizes[3] (sorted => du.Wl 16*64=1024, insertion => um.Wl 32*64=2048)
    bool sorted_keys = (in_sizes[3] == 1024);
    LayerPtr LP[3];
    for (int l = 0; l < 3; ++l) {
        int p = 3 + l * 19;
        auto g = [&](int off) { return (const float*)d_in[p + off]; };
        if (sorted_keys) {
            LP[l].du_Wl = g(0);  LP[l].du_Wr = g(1);  LP[l].du_bl = g(2);
            LP[l].gat_Wdst = g(3); LP[l].gat_Wsrc = g(4);
            LP[l].gat_ad = g(5);  LP[l].gat_as = g(6); LP[l].gat_b = g(7);
            LP[l].gcn_W = g(8);  LP[l].gcn_b = g(9);
            LP[l].mu_Wl = g(10); LP[l].mu_Wr = g(11); LP[l].mu_bl = g(12);
            LP[l].ud_Wl = g(13); LP[l].ud_Wr = g(14); LP[l].ud_bl = g(15);
            LP[l].um_Wl = g(16); LP[l].um_Wr = g(17); LP[l].um_bl = g(18);
        } else {
            LP[l].um_Wl = g(0);  LP[l].um_bl = g(1);  LP[l].um_Wr = g(2);
            LP[l].mu_Wl = g(3);  LP[l].mu_bl = g(4);  LP[l].mu_Wr = g(5);
            LP[l].ud_Wl = g(6);  LP[l].ud_bl = g(7);  LP[l].ud_Wr = g(8);
            LP[l].du_Wl = g(9);  LP[l].du_bl = g(10); LP[l].du_Wr = g(11);
            LP[l].gat_Wsrc = g(12); LP[l].gat_Wdst = g(13);
            LP[l].gat_as = g(14); LP[l].gat_ad = g(15); LP[l].gat_b = g(16);
            LP[l].gcn_W = g(17); LP[l].gcn_b = g(18);
        }
    }
    const int* e_trans = (const int*)d_in[60];   // src, dst=+E
    const int* e_recv  = (const int*)d_in[61];
    const int* e_uses  = (const int*)d_in[62];
    const int* e_usedb = (const int*)d_in[63];
    const int* e_temp  = (const int*)d_in[64];
    const int* e_sim   = (const int*)d_in[65];

    // ---- workspace carve ----
    char* ws = (char*)d_ws;
    size_t off = 0;
    auto alloc = [&](size_t bytes) -> void* {
        off = (off + 255) & ~(size_t)255;
        void* p = ws + off; off += bytes; return p;
    };
    float* xuA = (float*)alloc((size_t)NU * 64 * 4);
    float* xuB = (float*)alloc((size_t)NU * 64 * 4);
    float* xmA = (float*)alloc((size_t)NM * 64 * 4);
    float* xmB = (float*)alloc((size_t)NM * 64 * 4);
    float* xdA = (float*)alloc((size_t)ND * 64 * 4);
    float* xdB = (float*)alloc((size_t)ND * 64 * 4);
    float* agg = (float*)alloc((size_t)NU * 64 * 4);
    float* cnt = (float*)alloc((size_t)NU * 4);
    unsigned short* bfA   = (unsigned short*)alloc((size_t)NU * 64 * 2);
    unsigned short* bfB   = (unsigned short*)alloc((size_t)NU * 64 * 2);
    unsigned short* xu_bf = (unsigned short*)alloc((size_t)NU * 64 * 2);
    unsigned short* hs_bf = (unsigned short*)alloc((size_t)NU * 256 * 2);
    float*    a_s    = (float*)alloc((size_t)NU * 4 * 4);
    float*    a_d    = (float*)alloc((size_t)NU * 4 * 4);
    float*    denom  = (float*)alloc((size_t)NU * 4 * 4);
    unsigned* mmax   = (unsigned*)alloc((size_t)NU * 4 * 4);
    float*    walpha = (float*)alloc((size_t)E * 4 * 4);
    float*    stats  = (float*)alloc(128 * 4);
    float*    ws_eff = (float*)alloc(64 * 4 * 4);
    float*    wd_eff = (float*)alloc(64 * 4 * 4);
    unsigned short* wt_a = (unsigned short*)alloc((size_t)256 * 64 * 2);
    unsigned short* wt_b = (unsigned short*)alloc((size_t)256 * 64 * 2);

    float* out_u = (float*)d_out;
    float* out_m = out_u + (size_t)NU * 64;
    float* out_d = out_m + (size_t)NM * 64;

    // ---- host helpers ----
    auto gemm = [&](const unsigned short* A, const unsigned short* Bt, const float* bias,
                    float* Cf, unsigned short* Cbf, int M, int N, int KP, int acc) {
        dim3 grid(gdiv(M / 16, 8), N / 16);
        if (KP == 32)
            gemm_bf16_wmma<32><<<grid, 256, 0, stream>>>(A, Bt, bias, Cf, Cbf, M, N, acc);
        else
            gemm_bf16_wmma<64><<<grid, 256, 0, stream>>>(A, Bt, bias, Cf, Cbf, M, N, acc);
    };
    auto tpad = [&](const float* W, unsigned short* Wt, int K, int N, int KP) {
        transpose_pad_bf16_k<<<gdiv((size_t)N * KP, 256), 256, 0, stream>>>(W, Wt, K, N, KP);
    };
    auto cpad = [&](const float* x, unsigned short* y, int K, int KP, int rows) {
        to_bf16_pad_k<<<gdiv((size_t)rows * KP, 256), 256, 0, stream>>>(x, y, K, KP, (size_t)rows * KP);
    };
    auto sage = [&](const float* xsrc, int Ksrc, const unsigned short* xdst_bf,
                    int Ndst, int Kdst, const int* ei,
                    const float* Wl, const float* bl, const float* Wr,
                    float* acc, int accflag) {
        int kps = kpad(Ksrc), kpd = kpad(Kdst);
        tpad(Wl, wt_a, Ksrc, 64, kps);
        tpad(Wr, wt_b, Kdst, 64, kpd);
        fill_f32_k<<<gdiv((size_t)Ndst * Ksrc, 256), 256, 0, stream>>>(agg, 0.f, (size_t)Ndst * Ksrc);
        fill_f32_k<<<gdiv(Ndst, 256), 256, 0, stream>>>(cnt, 0.f, Ndst);
        seg_sum_k<<<gdiv((size_t)E * Ksrc, 256), 256, 0, stream>>>(xsrc, ei, ei + E, agg, cnt, E, Ksrc);
        seg_div_k<<<gdiv((size_t)Ndst * Ksrc, 256), 256, 0, stream>>>(agg, cnt, Ndst, Ksrc);
        cpad(agg, bfA, Ksrc, kps, Ndst);
        gemm(bfA, wt_a, bl, acc, nullptr, Ndst, 64, kps, accflag);
        gemm(xdst_bf, wt_b, nullptr, acc, nullptr, Ndst, 64, kpd, 1);
    };
    auto bn = [&](const float* x, float* out, int N) {
        fill_f32_k<<<1, 128, 0, stream>>>(stats, 0.f, 128);
        bn_reduce_k<<<512, 256, 0, stream>>>(x, stats, N);
        bn_apply_k<<<gdiv((size_t)N * 64, 256), 256, 0, stream>>>(x, stats, out, N);
    };

    // ---- 3 layers ----
    const float* cu = x_user;     int du = 32;
    const float* cm = x_merchant; int dm = 16;
    const float* cd = x_device;   int dd = 16;
    float* accU[3] = { xuA, xuB, xuA };
    float* accM[3] = { xmA, xmB, xmA };
    float* accD[3] = { xdA, xdB, xdA };
    float* bnU[3]  = { xuA, xuB, out_u };
    float* bnM[3]  = { xmA, xmB, out_m };
    float* bnD[3]  = { xdA, xdB, out_d };

    for (int l = 0; l < 3; ++l) {
        const LayerPtr& P = LP[l];
        float* au = accU[l];
        float* am = accM[l];
        float* ad = accD[l];
        int kpu = kpad(du);

        // bf16 copy (K-padded) of current user features, reused by all user-side GEMMs
        cpad(cu, xu_bf, du, kpu, NU);

        // --- user: SAGE merchant->user (overwrite + bias), SAGE device->user ---
        sage(cm, dm, xu_bf, NU, du, e_recv,  P.mu_Wl, P.mu_bl, P.mu_Wr, au, 0);
        sage(cd, dd, xu_bf, NU, du, e_usedb, P.du_Wl, P.du_bl, P.du_Wr, au, 1);

        // --- user: GCN over "similar" ---
        fill_f32_k<<<gdiv(NU, 256), 256, 0, stream>>>(cnt, 1.f, NU);      // deg incl. self loop
        deg_count_k<<<gdiv(E, 256), 256, 0, stream>>>(e_sim + E, cnt, E);
        tpad(P.gcn_W, wt_a, du, 64, kpu);
        gemm(xu_bf, wt_a, nullptr, agg, nullptr, NU, 64, kpu, 0);         // h = x @ W
        gcn_edge_k<<<gdiv((size_t)E * 64, 256), 256, 0, stream>>>(agg, cnt, e_sim, e_sim + E, au, E);
        gcn_self_k<<<gdiv(NU, 256), 256, 0, stream>>>(agg, cnt, P.gcn_b, au, NU);

        // --- user: GAT over "temporal" ---
        att_eff_k<<<gdiv(du * 4, 128), 128, 0, stream>>>(P.gat_Wsrc, P.gat_as, ws_eff, du);
        att_eff_k<<<gdiv(du * 4, 128), 128, 0, stream>>>(P.gat_Wdst, P.gat_ad, wd_eff, du);
        gat_scores_k<<<gdiv(NU, 256), 256, 0, stream>>>(cu, ws_eff, wd_eff, a_s, a_d, NU, du);
        tpad(P.gat_Wsrc, wt_a, du, 256, kpu);
        gemm(xu_bf, wt_a, nullptr, nullptr, hs_bf, NU, 256, kpu, 0);      // hs in bf16
        fill_u32_k<<<gdiv((size_t)NU * 4, 256), 256, 0, stream>>>(mmax, 0x007FFFFFu, (size_t)NU * 4); // fmap(-inf)
        gat_edge_max_k<<<gdiv(E, 256), 256, 0, stream>>>(a_s, a_d, e_temp, e_temp + E, mmax, E);
        gat_self_max_k<<<gdiv(NU, 256), 256, 0, stream>>>(a_s, a_d, mmax, NU);
        fill_f32_k<<<gdiv((size_t)NU * 4, 256), 256, 0, stream>>>(denom, 0.f, (size_t)NU * 4);
        gat_edge_sum_k<<<gdiv(E, 256), 256, 0, stream>>>(a_s, a_d, e_temp, e_temp + E, mmax, denom, E);
        gat_self_sum_k<<<gdiv(NU, 256), 256, 0, stream>>>(a_s, a_d, mmax, denom, NU);
        gat_edge_w_k<<<gdiv(E, 256), 256, 0, stream>>>(a_s, a_d, e_temp, e_temp + E, mmax, denom, walpha, E);
        gat_edge_acc_k<<<gdiv((size_t)E * 64, 256), 256, 0, stream>>>(walpha, e_temp, e_temp + E, hs_bf, au, E);
        gat_self_acc_k<<<gdiv(NU, 256), 256, 0, stream>>>(a_s, a_d, mmax, denom, hs_bf, P.gat_b, au, NU);

        // --- merchant: SAGE user->merchant ---
        cpad(cm, bfB, dm, kpad(dm), NM);
        sage(cu, du, bfB, NM, dm, e_trans, P.um_Wl, P.um_bl, P.um_Wr, am, 0);

        // --- device: SAGE user->device ---
        cpad(cd, bfB, dd, kpad(dd), ND);
        sage(cu, du, bfB, ND, dd, e_uses, P.ud_Wl, P.ud_bl, P.ud_Wr, ad, 0);

        // --- BatchNorm + ReLU ---
        bn(au, bnU[l], NU);
        bn(am, bnM[l], NM);
        bn(ad, bnD[l], ND);

        cu = bnU[l]; cm = bnM[l]; cd = bnD[l];
        du = dm = dd = 64;
    }
}
